// FastWeightMemory_74543452389640
// MI455X (gfx1250) — compile-verified, matrix operand order checked by reading, not run on hardware
//
#include <hip/hip_runtime.h>

typedef __attribute__((ext_vector_type(16))) __bf16 v16bf;
typedef __attribute__((ext_vector_type(8)))  float  v8f;
typedef __attribute__((ext_vector_type(4)))  int    v4i;

union B4 { __bf16 b[4]; uint2 u; };
union B8 { __bf16 b[8]; uint4 u; };

// ---------- async Global->LDS copy (CDNA5 GLOBAL_LOAD_ASYNC_TO_LDS_B128) ----
#if defined(__AMDGCN__) && __has_builtin(__builtin_amdgcn_global_load_async_to_lds_b128)
#define HAVE_ASYNC_LDS 1
#else
#define HAVE_ASYNC_LDS 0
#endif

__device__ __forceinline__ void copy_b128_to_lds(const void* g, void* lds) {
#if HAVE_ASYNC_LDS
  typedef __attribute__((address_space(1))) v4i* gp;
  typedef __attribute__((address_space(3))) v4i* lp;
  __builtin_amdgcn_global_load_async_to_lds_b128((gp)g, (lp)lds, 0, 0);
#else
  *(uint4*)lds = *(const uint4*)g;
#endif
}

__device__ __forceinline__ void async_copy_wait() {
#if HAVE_ASYNC_LDS
#if __has_builtin(__builtin_amdgcn_s_wait_asynccnt)
  __builtin_amdgcn_s_wait_asynccnt(0);
#else
  asm volatile("s_wait_asynccnt 0" ::: "memory");
#endif
#endif
}

// ---------- WMMA fragment load from an LDS tile of [rows][64] bf16 ----------
// 16-bit A-matrix 16x32 layout (ISA 7.12.2): lane m=L&15, half h=L>>4.
// VGPR v holds K pair starting at (v>>2)*16 + h*8 + (v&3)*2, i.e. row dwords
// {h*4..h*4+3, 8+h*4..8+h*4+3} (+16 dwords for the second K=32 step).
// B fragments use the same loader on an [n][k]-transposed LDS tile.
__device__ __forceinline__ v16bf load_frag64(const __bf16* tile, int row, int ks) {
  const int lane = threadIdx.x & 31;
  const int m = lane & 15;
  const int h = lane >> 4;
  const unsigned int* p = (const unsigned int*)(tile + (row + m) * 64) + ks * 16;
  union { unsigned int u[8]; v16bf v; } r;
#pragma unroll
  for (int t = 0; t < 4; ++t) {
    r.u[t]     = p[h * 4 + t];
    r.u[4 + t] = p[8 + h * 4 + t];
  }
  return r.v;
}

// ---------- one K=64 stage of a 128x128 block tile, 8 waves (4M x 2N) ----------
__device__ __forceinline__ void wave_mma_stage(const __bf16* As, const __bf16* Bs,
                                               v8f (&acc)[2][4]) {
  const int w  = threadIdx.x >> 5;
  const int wm = (w & 3) * 32;   // wave row offset
  const int wn = (w >> 2) * 64;  // wave col offset
#pragma unroll
  for (int ks = 0; ks < 2; ++ks) {
    v16bf a[2], b[4];
#pragma unroll
    for (int mi = 0; mi < 2; ++mi) a[mi] = load_frag64(As, wm + mi * 16, ks);
#pragma unroll
    for (int ni = 0; ni < 4; ++ni) b[ni] = load_frag64(Bs, wn + ni * 16, ks);
#pragma unroll
    for (int mi = 0; mi < 2; ++mi)
#pragma unroll
      for (int ni = 0; ni < 4; ++ni)
        acc[mi][ni] = __builtin_amdgcn_wmma_f32_16x16x32_bf16(
            false, a[mi], false, b[ni], (short)0, acc[mi][ni], false, false);
  }
}

#define EPILOGUE_VARS()                           \
  const int lane = threadIdx.x & 31;              \
  const int w = threadIdx.x >> 5;                 \
  const int wm = (w & 3) * 32, wn = (w >> 2) * 64;\
  const int ccol = lane & 15, rb = (lane >> 4) * 8

#define ZERO_ACC()                                \
  v8f z = {};                                     \
  v8f acc[2][4];                                  \
  _Pragma("unroll")                               \
  for (int mi = 0; mi < 2; ++mi)                  \
    _Pragma("unroll")                             \
    for (int ni = 0; ni < 4; ++ni) acc[mi][ni] = z

// =======================================================================
// Kernel 1: fused QKV projection. out_g = bf16( X(32768x1024) @ W_g(1024x256) )
// grid = (256 mblocks * 2 nblocks, 3), block = 256
// =======================================================================
__global__ __launch_bounds__(256) void proj_kernel(
    const float* __restrict__ x,
    const float* __restrict__ Wq, const float* __restrict__ Wk, const float* __restrict__ Wv,
    __bf16* __restrict__ q, __bf16* __restrict__ k, __bf16* __restrict__ v) {
  __shared__ __bf16 As[128 * 64];
  __shared__ __bf16 Bs[128 * 64];
  const float* W   = (blockIdx.y == 0) ? Wq : (blockIdx.y == 1 ? Wk : Wv);
  __bf16*      out = (blockIdx.y == 0) ? q : (blockIdx.y == 1 ? k : v);
  const int mb = (int)(blockIdx.x >> 1) * 128;
  const int nb = (int)(blockIdx.x & 1) * 128;
  const int tid = threadIdx.x;
  ZERO_ACC();

  for (int kb = 0; kb < 1024; kb += 64) {
    __syncthreads();
#pragma unroll
    for (int l = 0; l < 8; ++l) {
      int qi = tid + 256 * l;
      // A: x rows, float4-vectorized along K
      {
        int r = qi >> 4, c = (qi & 15) * 4;
        const float* src = &x[(size_t)(mb + r) * 1024 + kb + c];
        float4 f = *(const float4*)src;
        if (kb + 64 < 1024) __builtin_prefetch(src + 64, 0, 0);  // next K tile
        B4 t; t.b[0] = (__bf16)f.x; t.b[1] = (__bf16)f.y;
              t.b[2] = (__bf16)f.z; t.b[3] = (__bf16)f.w;
        *(uint2*)(As + r * 64 + c) = t.u;
      }
      // B: W staged transposed Bs[n][k], float4-vectorized along N
      {
        int kk = qi >> 5, n4 = (qi & 31) * 4;
        float4 f = *(const float4*)&W[(size_t)(kb + kk) * 256 + nb + n4];
        Bs[(n4 + 0) * 64 + kk] = (__bf16)f.x;
        Bs[(n4 + 1) * 64 + kk] = (__bf16)f.y;
        Bs[(n4 + 2) * 64 + kk] = (__bf16)f.z;
        Bs[(n4 + 3) * 64 + kk] = (__bf16)f.w;
      }
    }
    __syncthreads();
    wave_mma_stage(As, Bs, acc);
  }

  EPILOGUE_VARS();
#pragma unroll
  for (int mi = 0; mi < 2; ++mi)
#pragma unroll
    for (int ni = 0; ni < 4; ++ni)
#pragma unroll
      for (int j = 0; j < 8; ++j) {
        int row = mb + wm + mi * 16 + rb + j;
        int col = nb + wn + ni * 16 + ccol;
        out[(size_t)row * 256 + col] = (__bf16)acc[mi][ni][j];
      }
}

// =======================================================================
// Kernel 2: in-place row L2 normalize of k and v (rows of 256 bf16).
// One wave per row; grid = (32768/8, 2), block = 256.
// =======================================================================
__global__ __launch_bounds__(256) void norm_kernel(__bf16* __restrict__ k,
                                                   __bf16* __restrict__ v) {
  __bf16* buf = (blockIdx.y == 0) ? k : v;
  const int lane = threadIdx.x & 31;
  const int w = threadIdx.x >> 5;
  const size_t row = (size_t)blockIdx.x * 8 + w;
  __bf16* p = buf + row * 256;
  B8 d; d.u = *(const uint4*)(p + lane * 8);   // 8 bf16 per lane, 16B coalesced
  float vals[8];
  float ss = 0.f;
#pragma unroll
  for (int e = 0; e < 8; ++e) {
    vals[e] = (float)d.b[e];
    ss += vals[e] * vals[e];
  }
#pragma unroll
  for (int off = 16; off > 0; off >>= 1) ss += __shfl_xor(ss, off, 32);
  float scale = 1.0f / fmaxf(sqrtf(ss), 1e-12f);
#pragma unroll
  for (int e = 0; e < 8; ++e) d.b[e] = (__bf16)(vals[e] * scale);
  *(uint4*)(p + lane * 8) = d.u;
}

// =======================================================================
// Kernel 3: per-chunk outer products O_t = (v_t^T @ k_t) / (B*c), fp32 out.
// M=N=256, K=512 (b,c enumeration). grid = (4, 64 chunks), block = 256.
// =======================================================================
__global__ __launch_bounds__(256) void outer_kernel(const __bf16* __restrict__ kmat,
                                                    const __bf16* __restrict__ vmat,
                                                    float* __restrict__ outer) {
  __shared__ __bf16 As[128 * 64];
  __shared__ __bf16 Bs[128 * 64];
  const int t = blockIdx.y;
  const int mb = (int)(blockIdx.x >> 1) * 128;  // i block (v dim)
  const int nb = (int)(blockIdx.x & 1) * 128;   // j block (k dim)
  const int tid = threadIdx.x;
  ZERO_ACC();

  for (int kb = 0; kb < 512; kb += 64) {
    __syncthreads();
#pragma unroll
    for (int l = 0; l < 4; ++l) {
      int gi = tid + 256 * l;
      int kk = gi >> 4, i8 = (gi & 15) * 8;      // 8 consecutive i per b128 load
      int kkg = kb + kk;                          // (b,c) index
      size_t grow = (size_t)(kkg >> 6) * 4096 + (size_t)t * 64 + (kkg & 63);
      B8 av; av.u = *(const uint4*)&vmat[grow * 256 + mb + i8];
      B8 bv; bv.u = *(const uint4*)&kmat[grow * 256 + nb + i8];
#pragma unroll
      for (int j = 0; j < 8; ++j) {
        As[(i8 + j) * 64 + kk] = av.b[j];        // A[i][kk] = v^T (transposed stage)
        Bs[(i8 + j) * 64 + kk] = bv.b[j];        // Bs[n=j][kk]
      }
    }
    __syncthreads();
    wave_mma_stage(As, Bs, acc);
  }

  const float inv_bc = 1.0f / 512.0f;
  EPILOGUE_VARS();
#pragma unroll
  for (int mi = 0; mi < 2; ++mi)
#pragma unroll
    for (int ni = 0; ni < 4; ++ni)
#pragma unroll
      for (int j = 0; j < 8; ++j) {
        int row = mb + wm + mi * 16 + rb + j;
        int col = nb + wn + ni * 16 + ccol;
        outer[(size_t)t * 65536 + (size_t)row * 256 + col] = acc[mi][ni][j] * inv_bc;
      }
}

// =======================================================================
// Kernel 4: elementwise decay prefix: M_t states (before chunk t), bf16 out.
// grid = 64, block = 256, 4 elements/thread, 64-step serial per element.
// =======================================================================
__global__ __launch_bounds__(256) void scan_kernel(const float* __restrict__ M0,
                                                   const float* __restrict__ outer,
                                                   __bf16* __restrict__ Mst,
                                                   float decay_c) {
  const size_t g4 = ((size_t)blockIdx.x * 256 + threadIdx.x) * 4;
  float4 m = *(const float4*)&M0[g4];
#pragma unroll 1
  for (int t = 0; t < 64; ++t) {
    B4 p;
    p.b[0] = (__bf16)m.x; p.b[1] = (__bf16)m.y;
    p.b[2] = (__bf16)m.z; p.b[3] = (__bf16)m.w;
    *(uint2*)&Mst[(size_t)t * 65536 + g4] = p.u;
    float4 o = *(const float4*)&outer[(size_t)t * 65536 + g4];
    m.x = decay_c * m.x + o.x;
    m.y = decay_c * m.y + o.y;
    m.z = decay_c * m.z + o.z;
    m.w = decay_c * m.w + o.w;
  }
}

// =======================================================================
// Kernel 5: retrievals r_t = q_t @ M_t^T, bf16 out.
// Per chunk: M=512 rows (b,c), N=256, K=256. grid = (8, 64), block = 256.
// M_t stored row-major == exactly the [n][k]-transposed staging B needs,
// so both tiles stage with async Global->LDS b128 copies (ASYNCcnt).
// =======================================================================
__global__ __launch_bounds__(256) void retrieve_kernel(const __bf16* __restrict__ q,
                                                       const __bf16* __restrict__ Mst,
                                                       __bf16* __restrict__ R) {
  __shared__ __bf16 As[128 * 64];
  __shared__ __bf16 Bs[128 * 64];
  const int t = blockIdx.y;
  const int mb = (int)(blockIdx.x >> 1) * 128;  // chunk-row block (0..511)
  const int nb = (int)(blockIdx.x & 1) * 128;   // j block
  const int tid = threadIdx.x;
  ZERO_ACC();

  for (int kb = 0; kb < 256; kb += 64) {
    __syncthreads();
#pragma unroll
    for (int l = 0; l < 4; ++l) {
      int gi = tid + 256 * l;
      int rl = gi >> 3, c8 = (gi & 7) * 8;
      int mg = mb + rl;
      size_t grow = (size_t)(mg >> 6) * 4096 + (size_t)t * 64 + (mg & 63);
      copy_b128_to_lds(&q[grow * 256 + kb + c8], As + rl * 64 + c8);
      copy_b128_to_lds(&Mst[(size_t)t * 65536 + (size_t)(nb + rl) * 256 + kb + c8],
                       Bs + rl * 64 + c8);
    }
    async_copy_wait();
    __syncthreads();
    wave_mma_stage(As, Bs, acc);
  }

  EPILOGUE_VARS();
#pragma unroll
  for (int mi = 0; mi < 2; ++mi)
#pragma unroll
    for (int ni = 0; ni < 4; ++ni)
#pragma unroll
      for (int j = 0; j < 8; ++j) {
        int mg = mb + wm + mi * 16 + rb + j;
        size_t grow = (size_t)(mg >> 6) * 4096 + (size_t)t * 64 + (mg & 63);
        int col = nb + wn + ni * 16 + ccol;
        R[grow * 256 + col] = (__bf16)acc[mi][ni][j];
      }
}

// =======================================================================
// Kernel 6: output projection out = R(32768x256 bf16) @ W_out(256x1024), fp32.
// grid = (256*8), block = 256. A tile staged via async Global->LDS.
// =======================================================================
__global__ __launch_bounds__(256) void out_kernel(const __bf16* __restrict__ R,
                                                  const float* __restrict__ Wout,
                                                  float* __restrict__ out) {
  __shared__ __bf16 As[128 * 64];
  __shared__ __bf16 Bs[128 * 64];
  const int mb = (int)(blockIdx.x >> 3) * 128;
  const int nb = (int)(blockIdx.x & 7) * 128;
  const int tid = threadIdx.x;
  ZERO_ACC();

  for (int kb = 0; kb < 256; kb += 64) {
    __syncthreads();
#pragma unroll
    for (int l = 0; l < 4; ++l) {
      int gi = tid + 256 * l;
      // A: R rows, pure async b128 copy
      int rl = gi >> 3, c8 = (gi & 7) * 8;
      copy_b128_to_lds(&R[(size_t)(mb + rl) * 256 + kb + c8], As + rl * 64 + c8);
    }
#pragma unroll
    for (int l = 0; l < 8; ++l) {
      int qi = tid + 256 * l;
      // B: W_out staged transposed, float4-vectorized along N
      int kk = qi >> 5, n4 = (qi & 31) * 4;
      const float* src = &Wout[(size_t)(kb + kk) * 1024 + nb + n4];
      float4 f = *(const float4*)src;
      if (kb + 64 < 256) __builtin_prefetch(src + (size_t)64 * 1024, 0, 0);
      Bs[(n4 + 0) * 64 + kk] = (__bf16)f.x;
      Bs[(n4 + 1) * 64 + kk] = (__bf16)f.y;
      Bs[(n4 + 2) * 64 + kk] = (__bf16)f.z;
      Bs[(n4 + 3) * 64 + kk] = (__bf16)f.w;
    }
    async_copy_wait();
    __syncthreads();
    wave_mma_stage(As, Bs, acc);
  }

  EPILOGUE_VARS();
#pragma unroll
  for (int mi = 0; mi < 2; ++mi)
#pragma unroll
    for (int ni = 0; ni < 4; ++ni)
#pragma unroll
      for (int j = 0; j < 8; ++j) {
        int row = mb + wm + mi * 16 + rb + j;
        int col = nb + wn + ni * 16 + ccol;
        out[(size_t)row * 1024 + col] = acc[mi][ni][j];
      }
}

// =======================================================================
extern "C" void kernel_launch(void* const* d_in, const int* in_sizes, int n_in,
                              void* d_out, int out_size, void* d_ws, size_t ws_size,
                              hipStream_t stream) {
  (void)in_sizes; (void)n_in; (void)out_size; (void)ws_size;
  const float* x    = (const float*)d_in[0];  // (8,4096,1024)
  const float* Wq   = (const float*)d_in[1];  // (1024,256)
  const float* Wk   = (const float*)d_in[2];
  const float* Wv   = (const float*)d_in[3];
  const float* Wout = (const float*)d_in[4];  // (256,1024)
  const float* M0   = (const float*)d_in[5];  // (256,256)
  // d_in[6] = chunk_size (= 64, compile-time constant here)

  char* ws = (char*)d_ws;
  const size_t MiB = 1024 * 1024;
  __bf16* q     = (__bf16*)(ws);                 // 16 MiB: 32768x256 bf16
  __bf16* k     = (__bf16*)(ws + 16 * MiB);      // 16 MiB
  __bf16* v     = (__bf16*)(ws + 32 * MiB);      // 16 MiB
  float*  outer = (float*) (ws + 48 * MiB);      // 16 MiB: 64x256x256 f32
  __bf16* Mst   = (__bf16*)(ws + 64 * MiB);      //  8 MiB: 64x256x256 bf16
  __bf16* R     = (__bf16*)(ws + 72 * MiB);      // 16 MiB
  const float decay_c = __builtin_powf(0.99f, 64.0f);

  proj_kernel<<<dim3(512, 3), 256, 0, stream>>>(x, Wq, Wk, Wv, q, k, v);
  norm_kernel<<<dim3(4096, 2), 256, 0, stream>>>(k, v);
  outer_kernel<<<dim3(4, 64), 256, 0, stream>>>(k, v, outer);
  scan_kernel<<<dim3(64), 256, 0, stream>>>(M0, outer, Mst, decay_c);
  retrieve_kernel<<<dim3(8, 64), 256, 0, stream>>>(q, Mst, R);
  out_kernel<<<dim3(2048), 256, 0, stream>>>(R, Wout, (float*)d_out);
}
